// EquivariantMessagePassing_45088566673913
// MI455X (gfx1250) — matrix operand order repaired and checked by reading, not compiled, not run on hardware
//
#include <hip/hip_runtime.h>

#define NN 100000
#define EE 1000000
#define FF 64
#define HH 128

typedef __attribute__((ext_vector_type(16))) _Float16 v16h;
typedef __attribute__((ext_vector_type(8)))  _Float16 v8h;
typedef __attribute__((ext_vector_type(4)))  _Float16 v4h;
typedef __attribute__((ext_vector_type(8)))  float    v8f;

__device__ __forceinline__ void lds_fence() {
  asm volatile("s_wait_dscnt 0" ::: "memory");
}

// A-fragment (16x32 f16): lane m=lane&15; elems 0..7 = K=32k+8*half.., elems 8..15 = K=32k+16+8*half..
__device__ __forceinline__ v16h ldsA(const _Float16* p) {
  v8h lo = *(const v8h*)p;
  v8h hi = *(const v8h*)(p + 16);
  v16h a;
#pragma unroll
  for (int i = 0; i < 8; ++i) { a[i] = lo[i]; a[8 + i] = hi[i]; }
  return a;
}
// B-fragment (32x16 f16): lane n=lane&15; elems 0..7 = K=32k+8*half.., elems 8..15 = K=32k+16+8*half.. ([N][K] LDS)
__device__ __forceinline__ v16h ldsB(const _Float16* p) {
  v8h lo = *(const v8h*)p;
  v8h hi = *(const v8h*)(p + 16);
  v16h b;
#pragma unroll
  for (int i = 0; i < 8; ++i) { b[i] = lo[i]; b[8 + i] = hi[i]; }
  return b;
}

__device__ __forceinline__ float silu_f(float v) {
  return v / (1.0f + __expf(-v));
}

__global__ void zero_out_kernel(float* __restrict__ p, long long n) {
  long long i = (long long)blockIdx.x * blockDim.x + threadIdx.x;
  long long stride = (long long)gridDim.x * blockDim.x;
  for (; i < n; i += stride) p[i] = 0.0f;
}

// LDS layout (bytes): WT1 34816 | WT2 17408 | WTp1 18432 | fp32 tables 2304 | 8 x 8960 per-wave
#define LDS_BYTES 144640

__global__ __launch_bounds__(256, 2) void egnn_fused_kernel(
    const float* __restrict__ x, const float* __restrict__ pos,
    const int* __restrict__ ei,
    const float* __restrict__ Wx1, const float* __restrict__ bx1,
    const float* __restrict__ Wx2, const float* __restrict__ bx2,
    const float* __restrict__ Wp1, const float* __restrict__ bp1,
    const float* __restrict__ Wp2, const float* __restrict__ bp2,
    float* __restrict__ outx, float* __restrict__ outp, int nTiles)
{
  extern __shared__ char smem[];
  _Float16* WT1  = (_Float16*)smem;            // 128(N) x 136(K), K contiguous
  _Float16* WT2  = WT1 + 128 * 136;            // 64(N)  x 136(K)
  _Float16* WTp1 = WT2 + 64 * 136;             // 128(N) x 72(K)
  float* bx1L  = (float*)(WTp1 + 128 * 72);
  float* w128L = bx1L + 128;                   // Wx1 row 128 (dist^2 coefficients)
  float* bp1L  = w128L + 128;
  float* wp2L  = bp1L + 128;
  float* bx2L  = wp2L + 128;
  char*  pwb   = (char*)(bx2L + 64);

  const int tid  = threadIdx.x;
  const int wave = tid >> 5;
  const int lane = tid & 31;
  const int half = lane >> 4;
  const int l15  = lane & 15;

  char* myPw   = pwb + (size_t)wave * 8960;
  _Float16* As = (_Float16*)myPw;              // 16 x 136 f16 (x_row | x_col)
  _Float16* Hs = As + 16 * 136;                // 16 x 136 f16 (SiLU hidden)
  int*   rowL  = (int*)(Hs + 16 * 136);
  int*   colL  = rowL + 16;
  float* distL = (float*)(colL + 16);
  float* wposL = distL + 16;

  // ---- stage weights once per block: fp32 -> f16, transposed to [N][K] ----
  for (int i = tid; i < 128 * 128; i += 256) {
    int n = i & 127, k = i >> 7;
    WT1[n * 136 + k] = (_Float16)Wx1[k * 128 + n];      // Wx1 is (129,128); rows 0..127
  }
  for (int i = tid; i < 64 * 128; i += 256) {
    int n = i & 63, k = i >> 6;
    WT2[n * 136 + k] = (_Float16)Wx2[k * 64 + n];       // Wx2 is (128,64)
  }
  for (int i = tid; i < 128 * 64; i += 256) {
    int n = i & 127, k = i >> 7;
    WTp1[n * 72 + k] = (_Float16)Wp1[k * 128 + n];      // Wp1 is (64,128)
  }
  if (tid < 128) {
    bx1L[tid]  = bx1[tid];
    w128L[tid] = Wx1[128 * 128 + tid];                  // last row of Wx1 (dist^2 feature)
    bp1L[tid]  = bp1[tid];
    wp2L[tid]  = Wp2[tid];                              // Wp2 is (128,1)
  }
  if (tid < 64) bx2L[tid] = bx2[tid];
  __syncthreads();

  const float bp2v = bp2[0];
  const int gridWaves = gridDim.x * 8;

  for (int tileIdx = blockIdx.x * 8 + wave; tileIdx < nTiles; tileIdx += gridWaves) {
    const int e0 = tileIdx * 16;

    // ---- per-edge staging: indices, rel_pos, dist^2 (lanes 0..15) ----
    float rx = 0.0f, ry = 0.0f, rz = 0.0f;
    int cSelf = -1;
    if (lane < 16) {
      int e = e0 + lane;
      bool valid = (e < EE);
      int ec = valid ? e : 0;
      int r = ei[ec];
      int c = ei[EE + ec];
      rowL[lane] = r;
      colL[lane] = valid ? c : -1;
      cSelf = valid ? c : -1;
      rx = pos[r * 3 + 0] - pos[c * 3 + 0];
      ry = pos[r * 3 + 1] - pos[c * 3 + 1];
      rz = pos[r * 3 + 2] - pos[c * 3 + 2];
      distL[lane] = valid ? (rx * rx + ry * ry + rz * rz) : 0.0f;
    }
    lds_fence();

    // ---- gather node features into LDS A-tile: half 0 -> x[row], half 1 -> x[col] ----
    {
      int idx = half ? colL[l15] : rowL[l15];
      if (idx < 0) idx = 0;
      const float4* src = (const float4*)(x + (size_t)idx * FF);
      _Float16* dst = As + l15 * 136 + half * 64;
#pragma unroll 4
      for (int j = 0; j < 16; ++j) {
        float4 v = src[j];
        v4h h4;
        h4[0] = (_Float16)v.x; h4[1] = (_Float16)v.y;
        h4[2] = (_Float16)v.z; h4[3] = (_Float16)v.w;
        *(v4h*)(dst + j * 4) = h4;
      }
    }
    lds_fence();

    float dloc[8];
    int   cloc[8];
#pragma unroll
    for (int r = 0; r < 8; ++r) {
      dloc[r] = distL[half * 8 + r];   // accumulator row m = 8*half + r
      cloc[r] = colL[half * 8 + r];
    }

    // ---- GEMM1: hidden = SiLU(feat @ Wx1[:128] + dist^2 * Wx1[128] + bx1) -> Hs (f16) ----
#pragma unroll
    for (int t = 0; t < 8; ++t) {
      v8f c = {};
#pragma unroll
      for (int k = 0; k < 4; ++k) {
        v16h a = ldsA(As + l15 * 136 + 32 * k + 8 * half);
        v16h b = ldsB(WT1 + (t * 16 + l15) * 136 + 32 * k + 8 * half);
        c = __builtin_amdgcn_wmma_f32_16x16x32_f16(false, a, false, b, (short)0, c, false, false);
      }
      const int n = t * 16 + l15;
      const float bn = bx1L[n], wn = w128L[n];
#pragma unroll
      for (int r = 0; r < 8; ++r) {
        float h = c[r] + bn + dloc[r] * wn;
        Hs[(half * 8 + r) * 136 + n] = (_Float16)silu_f(h);
      }
    }

    // ---- phi_pos: weight = SiLU(x_row @ Wp1 + bp1) . Wp2 + bp2 ----
    float wsum[8];
#pragma unroll
    for (int r = 0; r < 8; ++r) wsum[r] = 0.0f;
#pragma unroll
    for (int t = 0; t < 8; ++t) {
      v8f c = {};
#pragma unroll
      for (int k = 0; k < 2; ++k) {
        v16h a = ldsA(As + l15 * 136 + 32 * k + 8 * half);   // x_row = columns 0..63
        v16h b = ldsB(WTp1 + (t * 16 + l15) * 72 + 32 * k + 8 * half);
        c = __builtin_amdgcn_wmma_f32_16x16x32_f16(false, a, false, b, (short)0, c, false, false);
      }
      const int n = t * 16 + l15;
      const float bn = bp1L[n], wn = wp2L[n];
#pragma unroll
      for (int r = 0; r < 8; ++r) wsum[r] += silu_f(c[r] + bn) * wn;
    }
    // reduce across the 16 lanes of each half-wave (columns of the D fragment)
#pragma unroll
    for (int r = 0; r < 8; ++r) {
      float v = wsum[r];
      v += __shfl_xor(v, 1, 32);
      v += __shfl_xor(v, 2, 32);
      v += __shfl_xor(v, 4, 32);
      v += __shfl_xor(v, 8, 32);
      if (l15 == 0) wposL[half * 8 + r] = v + bp2v;
    }
    lds_fence();   // Hs + wposL visible to whole wave

    // ---- GEMM2: msg = SiLU_hidden @ Wx2 + bx2 ; scatter-add into aggregated_x ----
#pragma unroll
    for (int t = 0; t < 4; ++t) {
      v8f c = {};
#pragma unroll
      for (int k = 0; k < 4; ++k) {
        v16h a = ldsA(Hs + l15 * 136 + 32 * k + 8 * half);
        v16h b = ldsB(WT2 + (t * 16 + l15) * 136 + 32 * k + 8 * half);
        c = __builtin_amdgcn_wmma_f32_16x16x32_f16(false, a, false, b, (short)0, c, false, false);
      }
      const int n = t * 16 + l15;
      const float bn = bx2L[n];
#pragma unroll
      for (int r = 0; r < 8; ++r) {
        if (cloc[r] >= 0)
          atomicAdd(&outx[(size_t)cloc[r] * FF + n], c[r] + bn);
      }
    }

    // ---- pos update scatter: aggregated_pos[col] += weight * rel_pos ----
    if (lane < 16 && cSelf >= 0) {
      const float w = wposL[lane];
      atomicAdd(&outp[(size_t)cSelf * 3 + 0], w * rx);
      atomicAdd(&outp[(size_t)cSelf * 3 + 1], w * ry);
      atomicAdd(&outp[(size_t)cSelf * 3 + 2], w * rz);
    }
  }
}

extern "C" void kernel_launch(void* const* d_in, const int* in_sizes, int n_in,
                              void* d_out, int out_size, void* d_ws, size_t ws_size,
                              hipStream_t stream) {
  (void)in_sizes; (void)n_in; (void)d_ws; (void)ws_size; (void)out_size;

  const float* x   = (const float*)d_in[0];
  const float* pos = (const float*)d_in[1];
  const int*   ei  = (const int*)d_in[2];
  const float* Wx1 = (const float*)d_in[3];
  const float* bx1 = (const float*)d_in[4];
  const float* Wx2 = (const float*)d_in[5];
  const float* bx2 = (const float*)d_in[6];
  const float* Wp1 = (const float*)d_in[7];
  const float* bp1 = (const float*)d_in[8];
  const float* Wp2 = (const float*)d_in[9];
  const float* bp2 = (const float*)d_in[10];

  float* outx = (float*)d_out;
  float* outp = outx + (size_t)NN * FF;

  // zero the atomic accumulation buffers (d_out is poisoned once by harness)
  const long long total = (long long)NN * FF + (long long)NN * 3;
  zero_out_kernel<<<1024, 256, 0, stream>>>((float*)d_out, total);

  const int nTiles = (EE + 15) / 16;
  hipFuncSetAttribute((const void*)egnn_fused_kernel,
                      hipFuncAttributeMaxDynamicSharedMemorySize, LDS_BYTES);
  egnn_fused_kernel<<<1024, 256, LDS_BYTES, stream>>>(
      x, pos, ei, Wx1, bx1, Wx2, bx2, Wp1, bp1, Wp2, bp2, outx, outp, nTiles);
}
